// SingleGNN_70703751627452
// MI455X (gfx1250) — compile-verified, matrix operand order checked by reading, not run on hardware
//
#include <hip/hip_runtime.h>
#include <hip/hip_bf16.h>

typedef __attribute__((ext_vector_type(8)))  _Float16 v8h;
typedef __attribute__((ext_vector_type(16))) _Float16 v16h;
typedef __attribute__((ext_vector_type(8)))  float    v8f;
typedef __attribute__((ext_vector_type(4)))  int      v4i;

static constexpr int   kN   = 16384;
static constexpr int   kE   = 65536;
static constexpr int   kG   = 256;
static constexpr int   kD   = 64;
static constexpr int   kDE  = 32;
static constexpr int   kDD  = kD * kD;      // 4096
static constexpr float kEPS = 1e-5f;

// gfx1250 async global->LDS copy path (compile-safe detection)
#if defined(__has_builtin)
#if __has_builtin(__builtin_amdgcn_global_load_async_to_lds_b128) && \
    __has_builtin(__builtin_amdgcn_s_wait_asynccnt)
#define USE_ASYNC_LDS 1
#endif
#endif
#ifndef USE_ASYNC_LDS
#define USE_ASYNC_LDS 0
#endif

#if USE_ASYNC_LDS
typedef __attribute__((address_space(1))) v4i gas_v4i;   // global int4
typedef __attribute__((address_space(3))) v4i las_v4i;   // LDS int4
#endif

// ---------------------------------------------------------------------------
// helpers
// ---------------------------------------------------------------------------
__device__ __forceinline__ float fast_tanhf(float x) {
    // tanh(x) = sign(x) * (1 - e) / (1 + e),  e = 2^(-2*log2(e)*|x|)
    float e = __builtin_amdgcn_exp2f(-2.8853900817779268f * fabsf(x));
    float r = (1.0f - e) * __builtin_amdgcn_rcpf(1.0f + e);
    return copysignf(r, x);
}

__device__ __forceinline__ unsigned fkey(float x) {
    unsigned b = __float_as_uint(x);
    return (b & 0x80000000u) ? ~b : (b | 0x80000000u);
}
__device__ __forceinline__ float funkey(unsigned k) {
    unsigned b = (k & 0x80000000u) ? (k & 0x7FFFFFFFu) : ~k;
    return __uint_as_float(b);
}

// ---------------------------------------------------------------------------
// utility kernels
// ---------------------------------------------------------------------------
__global__ void zero_kernel(float* __restrict__ p, int n) {
    for (int i = blockIdx.x * blockDim.x + threadIdx.x; i < n;
         i += gridDim.x * blockDim.x)
        p[i] = 0.0f;
}

// conv_W [32][4096] f32  ->  Wt [4096][32] f16   (column-major for B tiles)
__global__ void transpose_w_kernel(const float* __restrict__ W,
                                   _Float16* __restrict__ Wt) {
    int idx = blockIdx.x * blockDim.x + threadIdx.x;      // < 4096*32
    if (idx >= kDD * kDE) return;
    int col = idx >> 5;
    int k   = idx & 31;
    Wt[idx] = (_Float16)W[(size_t)k * kDD + col];
}

// h_pre[n][c] = relu(x[n][:3] @ node_W + node_b)
__global__ void node_embed_kernel(const float* __restrict__ x,
                                  const float* __restrict__ W,
                                  const float* __restrict__ b,
                                  float* __restrict__ out) {
    int idx = blockIdx.x * blockDim.x + threadIdx.x;      // < N*64
    if (idx >= kN * kD) return;
    int n = idx >> 6, c = idx & 63;
    float a = b[c];
    a = fmaf(x[n * 3 + 0], W[0 * kD + c], a);
    a = fmaf(x[n * 3 + 1], W[1 * kD + c], a);
    a = fmaf(x[n * 3 + 2], W[2 * kD + c], a);
    out[idx] = fmaxf(a, 0.0f);
}

// e_pre[e][c] = relu(edge_attr[e][:2] @ edge_W + edge_b)
__global__ void edge_embed_kernel(const float* __restrict__ ea,
                                  const float* __restrict__ W,
                                  const float* __restrict__ b,
                                  float* __restrict__ out) {
    int idx = blockIdx.x * blockDim.x + threadIdx.x;      // < E*32
    if (idx >= kE * kDE) return;
    int e = idx >> 5, c = idx & 31;
    float a = b[c];
    a = fmaf(ea[e * 2 + 0], W[0 * kDE + c], a);
    a = fmaf(ea[e * 2 + 1], W[1 * kDE + c], a);
    out[idx] = fmaxf(a, 0.0f);
}

// per-channel sum / sumsq (biased batch stats) — striped partials + few atomics
__global__ void stats_kernel(const float* __restrict__ x, int rows, int cols,
                             float* __restrict__ sum, float* __restrict__ sumsq) {
    int t = blockIdx.x * blockDim.x + threadIdx.x;
    int total = gridDim.x * blockDim.x;
    int c = t % cols;
    int stripe = t / cols;
    int nstripes = total / cols;
    float s = 0.0f, q = 0.0f;
    for (int r = stripe; r < rows; r += nstripes) {
        float v = x[(size_t)r * cols + c];
        s += v;
        q = fmaf(v, v, q);
    }
    atomicAdd(&sum[c], s);
    atomicAdd(&sumsq[c], q);
}

// y = (x - mu) * rsqrt(var + eps) * gamma + beta ; optional f32 and f16 outputs
__global__ void bn_apply_kernel(const float* __restrict__ x,
                                float* __restrict__ y, _Float16* __restrict__ yh,
                                int rows, int cols, float invRows,
                                const float* __restrict__ sum,
                                const float* __restrict__ sumsq,
                                const float* __restrict__ gamma,
                                const float* __restrict__ beta) {
    int total = rows * cols;
    for (int idx = blockIdx.x * blockDim.x + threadIdx.x; idx < total;
         idx += gridDim.x * blockDim.x) {
        int c = idx % cols;
        float mu  = sum[c] * invRows;
        float var = sumsq[c] * invRows - mu * mu;
        float v = (x[idx] - mu) * rsqrtf(var + kEPS) * gamma[c] + beta[c];
        if (y)  y[idx]  = v;
        if (yh) yh[idx] = (_Float16)v;
    }
}

__global__ void degree_kernel(const int* __restrict__ dst, float* __restrict__ cnt) {
    int e = blockIdx.x * blockDim.x + threadIdx.x;
    if (e < kE) atomicAdd(&cnt[dst[e]], 1.0f);
}

// ---------------------------------------------------------------------------
// NNConv edge kernel: fused  w = act(e @ W + b)  with  msg = h[src] @ w,
// scatter-add into agg.  One wave handles 16 edges; 256 WMMAs per wave.
// Wt is cooperatively double-buffered through LDS by the whole block
// (async global->LDS on gfx1250 when available).
// ---------------------------------------------------------------------------
static constexpr int CH_I = 2;            // i-rows of the 64x64 weight per chunk
static constexpr int TPC  = CH_I * 4;     // 16-col tiles per chunk   = 8
static constexpr int RPC  = TPC * 16;     // Wt rows per chunk        = 128
static constexpr int LROW = 40;           // halves per LDS row (80 B, padded,
                                          // 20*col mod 64 -> conflict-free)

template <int ACT>   // 0 = tanh, 1 = relu
__global__ __launch_bounds__(256)
void nnconv_edge_kernel(const _Float16* __restrict__ eh,   // [E][32] f16 (post-BN)
                        const _Float16* __restrict__ Wt,   // [4096][32] f16 (col-major)
                        const float* __restrict__ bias,    // [4096]
                        const float* __restrict__ hin,     // [N][64] f32
                        const int* __restrict__ src,
                        const int* __restrict__ dst,
                        float* __restrict__ agg) {         // [N][64] f32 (zeroed)
    __shared__ __align__(16) _Float16 s_wt[2][RPC * LROW]; // 2 x 10 KB
    __shared__ float s_h[8][kD][16];                       // 32 KB: [wave][i][edge]

    const int tid  = threadIdx.x;
    const int wave = tid >> 5;
    const int lane = tid & 31;
    const int col  = lane & 15;              // N-column of a tile / A-row (edge)
    const int hi   = lane >> 4;              // lane half
    const int ka   = hi ? 8 : 0;             // K-chunk select (ISA f16 A/B layout)

    const int grp   = blockIdx.x * 8 + wave;
    const int ebase = grp * 16;

    // stage h[src] for this wave's 16 edges, transposed: s_h[wave][i][m]
    for (int idx = lane; idx < 16 * kD; idx += 32) {
        int mm = idx & 15;
        int ii = idx >> 4;
        s_h[wave][ii][mm] = hin[(size_t)src[ebase + mm] * kD + ii];
    }

    // cooperative Wt chunk staging (block-wide)
    auto stage_chunk = [&](int ch, int b) {
        const _Float16* gsrc = Wt + (size_t)ch * (RPC * kDE);
        _Float16* ld = &s_wt[b][0];
        for (int j = tid; j < RPC * 4; j += 256) {      // 16 B per transfer
            int row = j >> 2, q = j & 3;
#if USE_ASYNC_LDS
            __builtin_amdgcn_global_load_async_to_lds_b128(
                (gas_v4i*)(gsrc + row * kDE + q * 8),
                (las_v4i*)(ld + row * LROW + q * 8), 0, 0);
#else
            *(v8h*)(ld + row * LROW + q * 8) =
                *(const v8h*)(gsrc + row * kDE + q * 8);
#endif
        }
    };

    // A operand: 16 edges x 32 feats, f16 (ISA 16-bit A 16x32 layout)
    const _Float16* ea = eh + (size_t)(ebase + col) * kDE;
    v8h a0 = *(const v8h*)(ea + ka);
    v8h a1 = *(const v8h*)(ea + ka + 16);
    v16h A = __builtin_shufflevector(a0, a1, 0, 1, 2, 3, 4, 5, 6, 7,
                                             8, 9, 10, 11, 12, 13, 14, 15);
    v8f cz = {};

    float msg[4][8];
#pragma unroll
    for (int oq = 0; oq < 4; ++oq)
#pragma unroll
        for (int r = 0; r < 8; ++r) msg[oq][r] = 0.0f;

    stage_chunk(0, 0);
#if USE_ASYNC_LDS
    __builtin_amdgcn_s_wait_asynccnt(0);
#endif
    __syncthreads();

    const int nch = kD / CH_I;                       // 32 chunks
    for (int ch = 0; ch < nch; ++ch) {
        const int buf = ch & 1;
        if (ch + 1 < nch) stage_chunk(ch + 1, buf ^ 1);

#pragma unroll
        for (int ii2 = 0; ii2 < CH_I; ++ii2) {
            const int i = ch * CH_I + ii2;
            // h[src[edge M]][i] for this lane's 8 rows (M = hi*8 + r)
            float hv[8];
            const float* hp = &s_h[wave][i][hi * 8];
#pragma unroll
            for (int r = 0; r < 8; ++r) hv[r] = hp[r];

#pragma unroll
            for (int oq = 0; oq < 4; ++oq) {
                const int tl = ii2 * 4 + oq;                 // tile within chunk
                const _Float16* wp = &s_wt[buf][(tl * 16 + col) * LROW];
                v8h b0 = *(const v8h*)(wp + ka);
                v8h b1 = *(const v8h*)(wp + ka + 16);
                v16h B = __builtin_shufflevector(b0, b1, 0, 1, 2, 3, 4, 5, 6, 7,
                                                         8, 9, 10, 11, 12, 13, 14, 15);
                v8f c = __builtin_amdgcn_wmma_f32_16x16x32_f16(
                    false, A, false, B, (short)0, cz, false, false);
                float bb = bias[i * kD + oq * 16 + col];
#pragma unroll
                for (int r = 0; r < 8; ++r) {
                    float w = c[r] + bb;
                    w = (ACT == 0) ? fast_tanhf(w) : fmaxf(w, 0.0f);
                    msg[oq][r] = fmaf(hv[r], w, msg[oq][r]);
                }
            }
        }
#if USE_ASYNC_LDS
        __builtin_amdgcn_s_wait_asynccnt(0);
#endif
        __syncthreads();
    }

    // scatter messages: lane owns o = oq*16 + col for edges M = hi*8 + r
    int dsts[8];
#pragma unroll
    for (int r = 0; r < 8; ++r) dsts[r] = dst[ebase + hi * 8 + r];
#pragma unroll
    for (int oq = 0; oq < 4; ++oq)
#pragma unroll
        for (int r = 0; r < 8; ++r)
            atomicAdd(agg + (size_t)dsts[r] * kD + oq * 16 + col, msg[oq][r]);
}

// hpre = hin @ root + agg/max(cnt,1) + bias
__global__ void conv_finalize_kernel(const float* __restrict__ hin,
                                     const float* __restrict__ agg,
                                     const float* __restrict__ cnt,
                                     const float* __restrict__ root,   // [64][64]
                                     const float* __restrict__ cbias,  // [64]
                                     float* __restrict__ hpre) {
    int idx = blockIdx.x * blockDim.x + threadIdx.x;       // < N*64
    if (idx >= kN * kD) return;
    int n = idx >> 6, c = idx & 63;
    const float* hr = hin + (size_t)n * kD;
    float acc = 0.0f;
#pragma unroll 8
    for (int k = 0; k < kD; ++k) acc = fmaf(hr[k], root[k * kD + c], acc);
    float a = agg[idx] / fmaxf(cnt[n], 1.0f);
    hpre[idx] = acc + a + cbias[c];
}

// ---------------------------------------------------------------------------
// pooling
// ---------------------------------------------------------------------------
__global__ void pool_accum_kernel(const float* __restrict__ h,
                                  const int* __restrict__ batch,
                                  float* __restrict__ psum,
                                  unsigned* __restrict__ pmaxk,
                                  float* __restrict__ gcnt) {
    int idx = blockIdx.x * blockDim.x + threadIdx.x;       // < N*64
    if (idx >= kN * kD) return;
    int n = idx >> 6, c = idx & 63;
    int g = batch[n];
    float v = h[idx];
    atomicAdd(&psum[g * kD + c], v);
    atomicMax(&pmaxk[g * kD + c], fkey(v));
    if (c == 0) atomicAdd(&gcnt[g], 1.0f);
}

__global__ void pool_finalize_kernel(const float* __restrict__ psum,
                                     const unsigned* __restrict__ pmaxk,
                                     const float* __restrict__ gcnt,
                                     float* __restrict__ out) {  // [G][128]
    int idx = blockIdx.x * blockDim.x + threadIdx.x;       // < G*64
    if (idx >= kG * kD) return;
    int g = idx >> 6, c = idx & 63;
    float cnt = gcnt[g];
    out[g * (2 * kD) + c] = psum[idx] / fmaxf(cnt, 1.0f);
    unsigned k = pmaxk[idx];
    out[g * (2 * kD) + kD + c] = (k == 0u) ? 0.0f : funkey(k);
}

// ---------------------------------------------------------------------------
// launch
// ---------------------------------------------------------------------------
extern "C" void kernel_launch(void* const* d_in, const int* in_sizes, int n_in,
                              void* d_out, int out_size, void* d_ws, size_t ws_size,
                              hipStream_t stream) {
    const float* x         = (const float*)d_in[0];
    const int*   edge_idx  = (const int*)d_in[1];
    const float* edge_attr = (const float*)d_in[2];
    const int*   batch_vec = (const int*)d_in[3];
    const float* node_W    = (const float*)d_in[4];
    const float* node_b    = (const float*)d_in[5];
    const float* node_bn_g = (const float*)d_in[6];
    const float* node_bn_b = (const float*)d_in[7];
    const float* edge_W    = (const float*)d_in[8];
    const float* edge_b    = (const float*)d_in[9];
    const float* edge_bn_g = (const float*)d_in[10];
    const float* edge_bn_b = (const float*)d_in[11];
    const float* conv1_W   = (const float*)d_in[12];
    const float* conv1_b   = (const float*)d_in[13];
    const float* conv1_root= (const float*)d_in[14];
    const float* conv1_bias= (const float*)d_in[15];
    const float* bn1_g     = (const float*)d_in[16];
    const float* bn1_b     = (const float*)d_in[17];
    const float* conv2_W   = (const float*)d_in[18];
    const float* conv2_b   = (const float*)d_in[19];
    const float* conv2_root= (const float*)d_in[20];
    const float* conv2_bias= (const float*)d_in[21];
    const float* bn2_g     = (const float*)d_in[22];
    const float* bn2_b     = (const float*)d_in[23];

    const int* src = edge_idx;
    const int* dst = edge_idx + kE;

    // ---- workspace carve-up ----
    char* w = (char*)d_ws;
    _Float16* Wt1  = (_Float16*)w;  w += (size_t)kDD * kDE * 2;   // 256 KB
    _Float16* Wt2  = (_Float16*)w;  w += (size_t)kDD * kDE * 2;   // 256 KB
    _Float16* e0h  = (_Float16*)w;  w += (size_t)kE * kDE * 2;    // 4 MB
    float* hA      = (float*)w;     w += (size_t)kN * kD * 4;     // 4 MB
    float* hB      = (float*)w;     w += (size_t)kN * kD * 4;     // 4 MB
    float* hPre    = (float*)w;     w += (size_t)kN * kD * 4;     // 4 MB
    float* e0pre   = (float*)w;     w += (size_t)kE * kDE * 4;    // 8 MB
    // ---- zero-initialized region ----
    float* zbase   = (float*)w;
    float* agg     = (float*)w;     w += (size_t)kN * kD * 4;     // 4 MB
    float* cnt     = (float*)w;     w += (size_t)kN * 4;
    float* stats   = (float*)w;     w += 1024 * 4;
    float* psum    = (float*)w;     w += (size_t)kG * kD * 4;
    float* gcnt    = (float*)w;     w += (size_t)kG * 4;
    unsigned* pmaxk= (unsigned*)w;  w += (size_t)kG * kD * 4;     // key(-inf)>0 -> 0 == empty
    const int zcount = (int)(((float*)w) - zbase);

    float* sumN = stats +   0, *sqN = stats +  64;
    float* sumE = stats + 128, *sqE = stats + 160;
    float* sum1 = stats + 192, *sq1 = stats + 256;
    float* sum2 = stats + 320, *sq2 = stats + 384;

    // 0) zero agg/cnt/stats/pool buffers
    zero_kernel<<<1024, 256, 0, stream>>>(zbase, zcount);

    // 1) pre-transpose edge-net weights to f16 [4096][32]
    transpose_w_kernel<<<(kDD * kDE + 255) / 256, 256, 0, stream>>>(conv1_W, Wt1);
    transpose_w_kernel<<<(kDD * kDE + 255) / 256, 256, 0, stream>>>(conv2_W, Wt2);

    // 2) node embed + BN -> hA
    node_embed_kernel<<<(kN * kD + 255) / 256, 256, 0, stream>>>(x, node_W, node_b, hPre);
    stats_kernel<<<16, 256, 0, stream>>>(hPre, kN, kD, sumN, sqN);
    bn_apply_kernel<<<4096, 256, 0, stream>>>(hPre, hA, nullptr, kN, kD, 1.0f / kN,
                                              sumN, sqN, node_bn_g, node_bn_b);

    // 3) edge embed + BN -> e0h (f16, WMMA A operand)
    edge_embed_kernel<<<(kE * kDE + 255) / 256, 256, 0, stream>>>(edge_attr, edge_W, edge_b, e0pre);
    stats_kernel<<<16, 256, 0, stream>>>(e0pre, kE, kDE, sumE, sqE);
    bn_apply_kernel<<<4096, 256, 0, stream>>>(e0pre, nullptr, e0h, kE, kDE, 1.0f / kE,
                                              sumE, sqE, edge_bn_g, edge_bn_b);

    // 4) in-degrees (mean aggregation denominator, shared by both convs)
    degree_kernel<<<(kE + 255) / 256, 256, 0, stream>>>(dst, cnt);

    // 5) conv1 (tanh edge net) -> hB
    nnconv_edge_kernel<0><<<kE / 16 / 8, 256, 0, stream>>>(e0h, Wt1, conv1_b, hA, src, dst, agg);
    conv_finalize_kernel<<<(kN * kD + 255) / 256, 256, 0, stream>>>(hA, agg, cnt,
                                                                    conv1_root, conv1_bias, hPre);
    stats_kernel<<<16, 256, 0, stream>>>(hPre, kN, kD, sum1, sq1);
    bn_apply_kernel<<<4096, 256, 0, stream>>>(hPre, hB, nullptr, kN, kD, 1.0f / kN,
                                              sum1, sq1, bn1_g, bn1_b);

    // 6) conv2 (relu edge net) -> hA (reuse)
    zero_kernel<<<1024, 256, 0, stream>>>(agg, kN * kD);
    nnconv_edge_kernel<1><<<kE / 16 / 8, 256, 0, stream>>>(e0h, Wt2, conv2_b, hB, src, dst, agg);
    conv_finalize_kernel<<<(kN * kD + 255) / 256, 256, 0, stream>>>(hB, agg, cnt,
                                                                    conv2_root, conv2_bias, hPre);
    stats_kernel<<<16, 256, 0, stream>>>(hPre, kN, kD, sum2, sq2);
    bn_apply_kernel<<<4096, 256, 0, stream>>>(hPre, hA, nullptr, kN, kD, 1.0f / kN,
                                              sum2, sq2, bn2_g, bn2_b);

    // 7) global mean/max pool -> out [G, 128]
    pool_accum_kernel<<<(kN * kD + 255) / 256, 256, 0, stream>>>(hA, batch_vec, psum, pmaxk, gcnt);
    pool_finalize_kernel<<<(kG * kD + 255) / 256, 256, 0, stream>>>(psum, pmaxk, gcnt,
                                                                    (float*)d_out);
}
